// MultiHeadSelfAttention_11570641895618
// MI455X (gfx1250) — matrix operand order repaired and checked by reading, not compiled, not run on hardware
//
#include <hip/hip_runtime.h>

#define DEV __device__ __forceinline__

typedef __attribute__((ext_vector_type(4)))  __bf16 bf16x4;
typedef __attribute__((ext_vector_type(8)))  __bf16 bf16x8;
typedef __attribute__((ext_vector_type(16))) __bf16 bf16x16;
typedef __attribute__((ext_vector_type(8)))  float  f32x8;
typedef __attribute__((ext_vector_type(4)))  unsigned int u32x4;
typedef __attribute__((ext_vector_type(4)))  int i32x4;
typedef __attribute__((ext_vector_type(8)))  int i32x8;

// ---------------- problem constants ----------------
#define CB   8
#define CN   1024
#define CD   768
#define CH   12
#define CHD  64
#define CM   (CB * CN)          /* 8192 rows for projection GEMMs */

// ---------------- CDNA5 async global->LDS copy ----------------
DEV unsigned lds_off(const void* p) { return (unsigned)(uintptr_t)p; }

DEV void async_load_b128(unsigned lds_addr, const void* g) {
  asm volatile("global_load_async_to_lds_b128 %0, %1, off"
               :: "v"(lds_addr), "v"((unsigned long long)(uintptr_t)g)
               : "memory");
}
DEV void wait_async0() { asm volatile("s_wait_asynccnt 0x0" ::: "memory"); }

DEV void async_copy32B(void* lds_dst, const __bf16* g) {
  unsigned la = lds_off(lds_dst);
  async_load_b128(la,      g);
  async_load_b128(la + 16, g + 8);
}

// ---------------- CDNA5 Tensor Data Mover (TDM) ----------------
// Build a 2D D# (ISA 8.3/8.4) and issue TENSOR_LOAD_TO_LDS. data_size=2 bytes,
// pad_enable inserts LDS padding so the tile lands with a padded row stride.
// pad_icode: pad after 2^(code+1) DWORDs; pad_acode: pad (code+1) DWORDs.
DEV void tdm_load_2d(const void* gaddr, unsigned lds_addr,
                     unsigned tensor_d0, unsigned tensor_d1,
                     unsigned tile_d0, unsigned tile_d1,
                     unsigned stride_elems,
                     unsigned pad_icode, unsigned pad_acode) {
  unsigned long long ga = (unsigned long long)(uintptr_t)gaddr;
  u32x4 g0;
  g0[0] = 1u;                                         // count=1 (valid, user)
  g0[1] = lds_addr;                                   // lds_addr [63:32]
  g0[2] = (unsigned)(ga & 0xFFFFFFFFu);               // global_addr [95:64]
  g0[3] = (unsigned)((ga >> 32) & 0x01FFFFFFu)        // global_addr [120:96]
        | 0x80000000u;                                // type=2 ("image")
  i32x8 g1;
  g1[0] = (int)((1u << 16)                            // data_size = 2 bytes
              | (1u << 20)                            // pad_enable
              | (pad_icode << 22) | (pad_acode << 25));
  g1[1] = (int)(tensor_d0 << 16);                     // tensor_dim0 [79:48]
  g1[2] = (int)((tensor_d0 >> 16) | (tensor_d1 << 16));  // dim0 hi | dim1 lo
  g1[3] = (int)((tensor_d1 >> 16) | (tile_d0 << 16)); // dim1 hi | tile_dim0
  g1[4] = (int)(tile_d1 & 0xFFFFu);                   // tile_dim1 (tile_dim2=0)
  g1[5] = (int)stride_elems;                          // tensor_dim0_stride lo32
  g1[6] = 0;
  g1[7] = 0;
  i32x4 gz = {0, 0, 0, 0};
#if __clang_major__ >= 23
  i32x8 gz8 = {0, 0, 0, 0, 0, 0, 0, 0};
  __builtin_amdgcn_tensor_load_to_lds(g0, g1, gz, gz, gz8, 0);
#else
  __builtin_amdgcn_tensor_load_to_lds(g0, g1, gz, gz, 0);
#endif
}

// ---------------- WMMA helpers ----------------
DEV f32x8 wmma_bf16(bf16x16 a, bf16x16 b, f32x8 c) {
  return __builtin_amdgcn_wmma_f32_16x16x32_bf16(
      false, a, false, b, (short)0, c, false, false);
}

// A-fragment (16x32 bf16), row-major tile, 'stride' elements/row (ISA 7.12.2).
DEV bf16x16 load_frag_a(const __bf16* base, int stride) {
  int lane = threadIdx.x & 31;
  int m  = lane & 15;
  int kb = (lane >> 4) << 3;            // 0 or 8
  const __bf16* p = base + m * stride;
  bf16x8 lo = *(const bf16x8*)(p + kb);
  bf16x8 hi = *(const bf16x8*)(p + 16 + kb);
  bf16x16 r;
#pragma unroll
  for (int i = 0; i < 8; ++i) { r[i] = lo[i]; r[i + 8] = hi[i]; }
  return r;
}

// B-fragment (32x16 bf16) from B^T storage (rows = B columns, contiguous K).
DEV bf16x16 load_frag_bT(const __bf16* base, int stride) {
  int lane = threadIdx.x & 31;
  int n  = lane & 15;
  int kb = (lane >> 4) << 3;            // 0 or 8
  const __bf16* p = base + n * stride + kb;
  bf16x8 lo = *(const bf16x8*)(p);
  bf16x8 hi = *(const bf16x8*)(p + 16);
  bf16x16 r;
#pragma unroll
  for (int i = 0; i < 8; ++i) { r[i] = lo[i]; r[i + 8] = hi[i]; }
  return r;
}

// ---------------- kernel 1: f32 -> bf16 elementwise ----------------
__global__ __launch_bounds__(256) void cvt_bf16_kernel(
    const float* __restrict__ x, __bf16* __restrict__ xb, int n) {
  int i = (blockIdx.x * 256 + threadIdx.x) * 4;
  if (i < n) {
    float4 v = *(const float4*)(x + i);
    bf16x4 o;
    o[0] = (__bf16)v.x; o[1] = (__bf16)v.y;
    o[2] = (__bf16)v.z; o[3] = (__bf16)v.w;
    *(bf16x4*)(xb + i) = o;
  }
}

// ---------------- kernel 2: W[k][n] f32 -> Wt[n][k] bf16 ----------------
__global__ __launch_bounds__(256) void wt_kernel(
    const float* __restrict__ W, __bf16* __restrict__ Wt) {
  __shared__ float t[32][33];
  int bx = blockIdx.x, by = blockIdx.y;
  int x = threadIdx.x, y0 = threadIdx.y;     // 32 x 8
#pragma unroll
  for (int j = 0; j < 32; j += 8)
    t[y0 + j][x] = W[(size_t)(by * 32 + y0 + j) * CD + bx * 32 + x];
  __syncthreads();
#pragma unroll
  for (int j = 0; j < 32; j += 8)
    Wt[(size_t)(bx * 32 + y0 + j) * CD + by * 32 + x] = (__bf16)t[x][y0 + j];
}

// ---------------- kernel 3: tiled WMMA GEMM, TDM-fed, double-buffered ---
// MODE 0: bf16 out [B,H,N,64] (Q,K); MODE 1: bf16 out [B,H,64,N] (V);
// MODE 2: f32 out row-major [M,768] (final projection).
#define LDT 40   /* padded LDS row stride: 32 data + 8 pad (TDM pad 4 DWORDs) */
template <int MODE>
__global__ __launch_bounds__(256) void gemm_kernel(
    const __bf16* __restrict__ A, const __bf16* __restrict__ Bt,
    const float* __restrict__ bias, void* __restrict__ out) {
  __shared__ __align__(16) __bf16 As[2][128 * LDT];
  __shared__ __align__(16) __bf16 Bs[2][128 * LDT];
  int tid = threadIdx.x;
  int lane = tid & 31, wave = tid >> 5;
  int wm = wave >> 2, wn = wave & 3;         // 2x4 wave grid -> 128x128 tile
  int bm = blockIdx.y, bn = blockIdx.x;

  const __bf16* gA = A  + (size_t)(bm * 128) * CD;   // tile row 0
  const __bf16* gB = Bt + (size_t)(bn * 128) * CD;

  // TDM issue: one lane per block; DMA engine ignores EXEC. Tile = 128 rows
  // x 32 elems, row len 64B -> pad_icode 3 (16 DWORDs), pad 16B -> pad_acode 3.
  auto issue = [&](int k0, int buf) {
    tdm_load_2d(gA + k0, lds_off(As[buf]), CD, CM, 32, 128, CD, 3, 3);
    tdm_load_2d(gB + k0, lds_off(Bs[buf]), CD, CD, 32, 128, CD, 3, 3);
  };

  if (tid == 0) issue(0, 0);

  f32x8 acc[4][2] = {};
  int buf = 0;
  for (int k0 = 0; k0 < CD; k0 += 32, buf ^= 1) {
    if (tid == 0) __builtin_amdgcn_s_wait_tensorcnt(0);
    __syncthreads();                         // publish LDS + retire old reads
    if (k0 + 32 < CD && tid == 0) issue(k0 + 32, buf ^ 1);

    bf16x16 af[4], bfv[2];
#pragma unroll
    for (int fm = 0; fm < 4; ++fm)
      af[fm] = load_frag_a(As[buf] + (wm * 64 + fm * 16) * LDT, LDT);
#pragma unroll
    for (int fn = 0; fn < 2; ++fn)
      bfv[fn] = load_frag_bT(Bs[buf] + (wn * 32 + fn * 16) * LDT, LDT);
#pragma unroll
    for (int fm = 0; fm < 4; ++fm)
#pragma unroll
      for (int fn = 0; fn < 2; ++fn)
        acc[fm][fn] = wmma_bf16(af[fm], bfv[fn], acc[fm][fn]);
  }

  // epilogue: C-frag -> lanes 0-15: M=r, lanes 16-31: M=8+r; N=lane&15
  int colbase = bn * 128 + wn * 32;
  int rowtop  = bm * 128 + wm * 64 + ((lane >> 4) << 3);
#pragma unroll
  for (int fn = 0; fn < 2; ++fn) {
    int col = colbase + fn * 16 + (lane & 15);
    float bv = bias[col];
#pragma unroll
    for (int fm = 0; fm < 4; ++fm) {
      int rb = rowtop + fm * 16;
#pragma unroll
      for (int r = 0; r < 8; ++r) {
        float v  = acc[fm][fn][r] + bv;
        int row  = rb + r;
        if constexpr (MODE == 2) {
          ((float*)out)[(size_t)row * CD + col] = v;
        } else {
          int b = row >> 10, nr = row & (CN - 1);
          int h = col >> 6,  hd = col & (CHD - 1);
          size_t idx;
          if constexpr (MODE == 0)
            idx = ((size_t)(b * CH + h) * CN + nr) * CHD + hd;
          else
            idx = ((size_t)(b * CH + h) * CHD + hd) * CN + nr;
          ((__bf16*)out)[idx] = (__bf16)v;
        }
      }
    }
  }
}

// ---------------- kernel 4: flash attention (quiet softmax) -------------
// grid (16 q-tiles, 96 b*h), 128 threads (4 waves, 16 q-rows per wave).
// S^T = K@Q^T per 64-key tile (softmax reduce = per-lane + shfl_xor(16));
// ctx^T = V^T @ P^T; quiet denom = 1 + l. K/V tiles double-buffered via
// per-lane GLOBAL_LOAD_ASYNC_TO_LDS_B128 (ASYNCcnt path).
#define LKV 72   /* padded LDS stride for 64-wide tiles */
__global__ __launch_bounds__(128) void attn_kernel(
    const __bf16* __restrict__ Qb, const __bf16* __restrict__ Kb,
    const __bf16* __restrict__ Vt, __bf16* __restrict__ Cb) {
  __shared__ __align__(16) __bf16 Ks[2][64 * LKV];
  __shared__ __align__(16) __bf16 Vs[2][64 * LKV];
  __shared__ __align__(16) __bf16 Pw[4][16 * LKV];

  int tid = threadIdx.x, lane = tid & 31, wave = tid >> 5;
  int qt = blockIdx.x, bh = blockIdx.y;
  int qrow0 = qt * 64 + wave * 16;
  const size_t headNH = (size_t)bh * CN * CHD;    // [b,h] base for Q/K
  const size_t headHN = (size_t)bh * CHD * CN;    // [b,h] base for V^T

  // Q B-fragments (constant across key tiles); fold 1/sqrt(64)=2^-3 (exact
  // in bf16) so the softmax path needs no per-tile scaling.
  bf16x16 qb[2];
#pragma unroll
  for (int ks = 0; ks < 2; ++ks) {
    qb[ks] = load_frag_bT(Qb + headNH + (size_t)qrow0 * CHD + ks * 32, CHD);
#pragma unroll
    for (int i = 0; i < 16; ++i)
      qb[ks][i] = (__bf16)((float)qb[ks][i] * 0.125f);
  }

  int c0 = tid * 2;                               // 2 x 32B chunks per tile
  auto stage = [&](int kt, int buf) {
#pragma unroll
    for (int j = 0; j < 2; ++j) {
      int c = c0 + j;
      int row = c >> 2, seg = (c & 3) * 16;
      async_copy32B(Ks[buf] + row * LKV + seg,
                    Kb + headNH + (size_t)(kt * 64 + row) * CHD + seg);
      async_copy32B(Vs[buf] + row * LKV + seg,
                    Vt + headHN + (size_t)row * CN + kt * 64 + seg);
    }
  };

  stage(0, 0);
  f32x8 o[4] = {};
  float m_run = -1e30f, l_run = 0.f;
  int buf = 0;

  for (int kt = 0; kt < 16; ++kt, buf ^= 1) {
    wait_async0();
    __syncthreads();
    if (kt + 1 < 16) stage(kt + 1, buf ^ 1);      // overlap DMA with compute

    // S^T tile: 64 keys x 16 q  (A = K rows, B^T = Q rows)
    f32x8 s[4] = {};
#pragma unroll
    for (int ks = 0; ks < 2; ++ks)
#pragma unroll
      for (int fr = 0; fr < 4; ++fr) {
        bf16x16 ka = load_frag_a(Ks[buf] + fr * 16 * LKV + ks * 32, LKV);
        s[fr] = wmma_bf16(ka, qb[ks], s[fr]);
      }

    // online quiet softmax over keys (lane pair L, L+16 share q-column)
    float mt = -1e30f;
#pragma unroll
    for (int fr = 0; fr < 4; ++fr)
#pragma unroll
      for (int r = 0; r < 8; ++r) mt = fmaxf(mt, s[fr][r]);
    mt = fmaxf(mt, __shfl_xor(mt, 16, 32));
    float m_new = fmaxf(m_run, mt);
    float corr  = __expf(m_run - m_new);
    float lsum  = 0.f;
    int   qc = lane & 15, hiof = (lane >> 4) << 3;
#pragma unroll
    for (int fr = 0; fr < 4; ++fr) {
      bf16x8 pv;
#pragma unroll
      for (int r = 0; r < 8; ++r) {
        float e = __expf(s[fr][r] - m_new);
        lsum += e;
        pv[r] = (__bf16)e;
      }
      *(bf16x8*)(&Pw[wave][qc * LKV + fr * 16 + hiof]) = pv;  // P q-major
    }
    lsum += __shfl_xor(lsum, 16, 32);
    l_run = l_run * corr + lsum;
    m_run = m_new;
#pragma unroll
    for (int fm = 0; fm < 4; ++fm)
#pragma unroll
      for (int r = 0; r < 8; ++r) o[fm][r] *= corr;

    // ctx^T += V^T @ P^T  (A = V^T rows, B^T = P rows)
#pragma unroll
    for (int ks = 0; ks < 2; ++ks) {
      bf16x16 pb = load_frag_bT(&Pw[wave][ks * 32], LKV);
#pragma unroll
      for (int fm = 0; fm < 4; ++fm) {
        bf16x16 va = load_frag_a(Vs[buf] + fm * 16 * LKV + ks * 32, LKV);
        o[fm] = wmma_bf16(va, pb, o[fm]);
      }
    }
  }

  // epilogue: quiet denom = 1 + l ; ctx^T frag -> Cb[m, h*64+hd]
  float inv = 1.f / (1.f + l_run);
  int b = bh / CH, h = bh % CH;
  int qc = lane & 15, hiof = (lane >> 4) << 3;
  int row = qrow0 + qc;
  __bf16* dst = Cb + (size_t)(b * CN + row) * CD + h * CHD;
#pragma unroll
  for (int fm = 0; fm < 4; ++fm) {
    bf16x8 ov;
#pragma unroll
    for (int r = 0; r < 8; ++r) ov[r] = (__bf16)(o[fm][r] * inv);
    *(bf16x8*)(dst + fm * 16 + hiof) = ov;
  }
}

// ---------------- host-side launcher ----------------
extern "C" void kernel_launch(void* const* d_in, const int* in_sizes, int n_in,
                              void* d_out, int out_size, void* d_ws, size_t ws_size,
                              hipStream_t stream) {
  (void)in_sizes; (void)n_in; (void)out_size; (void)ws_size;
  const float* x  = (const float*)d_in[0];
  const float* Wq = (const float*)d_in[1];
  const float* bq = (const float*)d_in[2];
  const float* Wk = (const float*)d_in[3];
  const float* bk = (const float*)d_in[4];
  const float* Wv = (const float*)d_in[5];
  const float* bv = (const float*)d_in[6];
  const float* Wo = (const float*)d_in[7];
  const float* bo = (const float*)d_in[8];

  __bf16* ws = (__bf16*)d_ws;
  const size_t SZX = (size_t)CM * CD;        // 6291456
  const size_t SZW = (size_t)CD * CD;        // 589824
  __bf16* xb  = ws;                          // [8192,768]
  __bf16* Wt  = xb + SZX;                    // 4 x [768,768] transposed
  __bf16* Qb  = Wt + 4 * SZW;                // [B,H,N,64]
  __bf16* Kb  = Qb + SZX;                    // [B,H,N,64]
  __bf16* Vt  = Kb + SZX;                    // [B,H,64,N]
  __bf16* Cb  = Vt + SZX;                    // [8192,768] ctx

  cvt_bf16_kernel<<<(int)(SZX / 1024), 256, 0, stream>>>(x, xb, (int)SZX);
  dim3 tg(24, 24), tb(32, 8);
  wt_kernel<<<tg, tb, 0, stream>>>(Wq, Wt + 0 * SZW);
  wt_kernel<<<tg, tb, 0, stream>>>(Wk, Wt + 1 * SZW);
  wt_kernel<<<tg, tb, 0, stream>>>(Wv, Wt + 2 * SZW);
  wt_kernel<<<tg, tb, 0, stream>>>(Wo, Wt + 3 * SZW);

  dim3 gg(CD / 128, CM / 128);               // 6 x 64
  gemm_kernel<0><<<gg, 256, 0, stream>>>(xb, Wt + 0 * SZW, bq, Qb);
  gemm_kernel<0><<<gg, 256, 0, stream>>>(xb, Wt + 1 * SZW, bk, Kb);
  gemm_kernel<1><<<gg, 256, 0, stream>>>(xb, Wt + 2 * SZW, bv, Vt);

  attn_kernel<<<dim3(CN / 64, CB * CH), 128, 0, stream>>>(Qb, Kb, Vt, Cb);

  gemm_kernel<2><<<gg, 256, 0, stream>>>(Cb, Wt + 3 * SZW, bo, d_out);
}
